// TransformerMemoryNTM_45724221833218
// MI455X (gfx1250) — matrix operand order repaired and checked by reading, not compiled, not run on hardware
//
#include <hip/hip_runtime.h>
#include <hip/hip_bf16.h>
#include <math.h>

typedef __attribute__((ext_vector_type(16))) __bf16 v16bf;
typedef __attribute__((ext_vector_type(8)))  __bf16 v8bf;
typedef __attribute__((ext_vector_type(4)))  __bf16 v4bf;
typedef __attribute__((ext_vector_type(8)))  float  v8f;
typedef __attribute__((ext_vector_type(4)))  float  v4f;
typedef __attribute__((ext_vector_type(4)))  unsigned int v4u;
typedef __attribute__((ext_vector_type(8)))  int v8i;
typedef __attribute__((ext_vector_type(4)))  int v4i;

#define BB 64
#define LL 256
#define EE 512
#define HH 1024
#define NN 128
#define HS 64
#define NHD 4
#define OUTD 1024
#define INDIM 768
#define POD 1280
#define GK 1792      /* 768 + 1024 */
#define G4H 4096

/* ---------------- workspace layout (bytes) ---------------- */
#define OFF_H     ((size_t)0)
#define OFF_C     ((size_t)262144)
#define OFF_MEM   ((size_t)524288)
#define OFF_RC    ((size_t)2621440)
#define OFF_HN    ((size_t)2686976)
#define OFF_INP   ((size_t)2949120)
#define OFF_GATES ((size_t)3145728)
#define OFF_RKN   ((size_t)4194304)
#define OFF_WKN   ((size_t)4259840)
#define OFF_EV    ((size_t)4325376)
#define OFF_WSG   ((size_t)4390912)
#define OFF_PO    ((size_t)4395008)
#define OFF_WBG   ((size_t)4722688)   /* bf16 [4096][1792] */
#define OFF_WBH   ((size_t)19402752)  /* bf16 [768][1024]  */
#define OFF_WBP   ((size_t)20975616)  /* bf16 [1024][1280] */
#define OFF_WBQ   ((size_t)23597056)  /* bf16 [192][64]    */
#define OFF_WBO   ((size_t)23621632)  /* bf16 [64][64]     */
#define OFF_WB1   ((size_t)23629824)  /* bf16 [128][64]    */
#define OFF_WB2   ((size_t)23646208)  /* bf16 [64][128]    */

__device__ __forceinline__ float sigm_(float x){ return 1.0f/(1.0f+__expf(-x)); }
__device__ __forceinline__ float gelu_(float x){ return 0.5f*x*(1.0f+erff(x*0.70710678118654752f)); }

__device__ __forceinline__ v8f zero8(){ v8f z; for(int i=0;i<8;i++) z[i]=0.0f; return z; }

__device__ __forceinline__ v8f wmma_bf16(v16bf a, v16bf b, v8f c){
  return __builtin_amdgcn_wmma_f32_16x16x32_bf16(false, a, false, b, (short)0, c, false, false);
}

/* A fragment: 16x32 bf16 tile from row-major [M][lda] buffer.
   lane l (l<16): m=l, K = {0..7, 16..23}; lanes 16..31: K += 8. */
__device__ __forceinline__ v16bf frag_a(const __bf16* A, int lda, int m0, int k0, int lane){
  const __bf16* p = A + (size_t)(m0 + (lane & 15)) * lda + k0 + ((lane >> 4) * 8);
  v16bf f;
  #pragma unroll
  for(int i=0;i<8;i++){ f[i] = p[i]; f[i+8] = p[16+i]; }
  return f;
}

/* B fragment: 32x16 bf16 tile; B stored transposed as [N][ldb] row-major (i.e. W[n][k]). */
__device__ __forceinline__ v16bf frag_b(const __bf16* Bt, int ldb, int n0, int k0, int lane){
  const __bf16* p = Bt + (size_t)(n0 + (lane & 15)) * ldb + k0 + ((lane >> 4) * 8);
  v16bf f;
  #pragma unroll
  for(int i=0;i<8;i++){ f[i] = p[i]; f[i+8] = p[16+i]; }
  return f;
}

/* stage a [64 x 32] f32 tile (uniform source) into LDS as bf16, 8 elems/thread */
__device__ __forceinline__ void stage_a64x32(const float* src, int sld, int scol,
                                             __bf16* As, int tid){
  int row = tid >> 2, col = (tid & 3) * 8;
  const float* p = src + (size_t)row * sld + scol + col;
  v4f f0 = *(const v4f*)p;
  v4f f1 = *(const v4f*)(p + 4);
  v8bf o;
  o[0]=(__bf16)f0.x; o[1]=(__bf16)f0.y; o[2]=(__bf16)f0.z; o[3]=(__bf16)f0.w;
  o[4]=(__bf16)f1.x; o[5]=(__bf16)f1.y; o[6]=(__bf16)f1.z; o[7]=(__bf16)f1.w;
  *(v8bf*)(&As[row*40 + col]) = o;
}

__device__ __forceinline__ void block_red2(float& s, float& q, float* sh){
  int t = threadIdx.x;
  sh[t] = s; sh[256 + t] = q; __syncthreads();
  for(int off = 128; off > 0; off >>= 1){
    if(t < off){ sh[t] += sh[t+off]; sh[256+t] += sh[256+t+off]; }
    __syncthreads();
  }
  s = sh[0]; q = sh[256]; __syncthreads();
}

/* ------------------- prep kernels ------------------- */
__global__ void kZero(float* p, int n){ int i = blockIdx.x*256 + threadIdx.x; if(i < n) p[i] = 0.0f; }

__global__ void kCvt(const float* s, __bf16* d, int n){
  int i = blockIdx.x*256 + threadIdx.x; if(i < n) d[i] = (__bf16)s[i];
}

/* 4 elems per thread; boundary (768) and row length (1792) are multiples of 4 */
__global__ void kCvtGates(const float* Wih, const float* Whh, __bf16* d){
  int i = (blockIdx.x*256 + threadIdx.x) * 4; if(i >= G4H*GK) return;
  int n = i / GK, k = i % GK;
  const float* p = (k < INDIM) ? (Wih + (size_t)n*INDIM + k) : (Whh + (size_t)n*HH + (k - INDIM));
  v4f f = *(const v4f*)p;
  v4bf o; o[0]=(__bf16)f.x; o[1]=(__bf16)f.y; o[2]=(__bf16)f.z; o[3]=(__bf16)f.w;
  *(v4bf*)(d + i) = o;
}

__global__ void kCvtHeads(const float* Wrk, const float* Wwk, const float* Wev, __bf16* d){
  int i = (blockIdx.x*256 + threadIdx.x) * 4; if(i >= 768*HH) return;
  int n = i / HH, k = i % HH;
  const float* s = (n < 256) ? (Wrk + (size_t)n*HH) : (n < 512) ? (Wwk + (size_t)(n-256)*HH) : (Wev + (size_t)(n-512)*HH);
  v4f f = *(const v4f*)(s + k);
  v4bf o; o[0]=(__bf16)f.x; o[1]=(__bf16)f.y; o[2]=(__bf16)f.z; o[3]=(__bf16)f.w;
  *(v4bf*)(d + i) = o;
}

/* ------------------- step kernels ------------------- */
/* kA: inp = LN(cat(x_t, rc)) */
__global__ __launch_bounds__(256) void kA(const float* x_emb, const float* rc,
                                          const float* in_g, const float* in_b,
                                          float* inp, int t){
  int b = blockIdx.x, tid = threadIdx.x;
  __shared__ float sh[512];
  float v[3]; float s = 0.0f, q = 0.0f;
  #pragma unroll
  for(int i=0;i<3;i++){
    int j = tid + i*256;
    float x = (j < EE) ? x_emb[((size_t)b*LL + t)*EE + j] : rc[b*256 + (j - EE)];
    v[i] = x; s += x; q += x*x;
  }
  block_red2(s, q, sh);
  float mu = s * (1.0f/768.0f);
  float rs = rsqrtf(q * (1.0f/768.0f) - mu*mu + 1e-5f);
  #pragma unroll
  for(int i=0;i<3;i++){ int j = tid + i*256; inp[b*INDIM + j] = (v[i]-mu)*rs*in_g[j] + in_b[j]; }
}

/* kB: gates = [inp|h] @ [W_ih|W_hh]^T + b   (M=64, N=4096, K=1792, bf16 WMMA) */
__global__ __launch_bounds__(256) void kB(const float* inp, const float* h, const __bf16* Wbg,
                                          const float* b_ih, const float* b_hh, float* gates){
  __shared__ __bf16 As[64*40];
  int tid = threadIdx.x, lane = tid & 31, w = tid >> 5;
  int m0 = (w & 3) * 16, nh = w >> 2;
  int ngBase = blockIdx.x * 64;
  v8f ca = zero8(), cb = zero8();
  for(int k0 = 0; k0 < GK; k0 += 32){
    /* uniform per-tile source: tiles never straddle the 768 boundary */
    if(k0 < INDIM) stage_a64x32(inp, INDIM, k0, As, tid);
    else           stage_a64x32(h,   HH,    k0 - INDIM, As, tid);
    __syncthreads();
    __builtin_prefetch(Wbg + (size_t)(ngBase + nh*32)*GK + k0 + 128, 0, 3);
    v16bf a  = frag_a(As, 40, m0, 0, lane);
    v16bf b0 = frag_b(Wbg, GK, ngBase + nh*32,      k0, lane);
    v16bf b1 = frag_b(Wbg, GK, ngBase + nh*32 + 16, k0, lane);
    ca = wmma_bf16(a, b0, ca);
    cb = wmma_bf16(a, b1, cb);
    __syncthreads();
  }
  int nA = ngBase + nh*32 + (lane & 15), nB = nA + 16;
  float bA = b_ih[nA] + b_hh[nA], bBv = b_ih[nB] + b_hh[nB];
  int mo = m0 + ((lane >> 4) << 3);
  #pragma unroll
  for(int r=0;r<8;r++){
    gates[(size_t)(mo+r)*G4H + nA] = ca[r] + bA;
    gates[(size_t)(mo+r)*G4H + nB] = cb[r] + bBv;
  }
}

/* kC: LSTM pointwise + hn = LN(h) + write-strength gate */
__global__ __launch_bounds__(256) void kC(const float* gates, float* c, float* h, float* hn,
                                          const float* cn_g, const float* cn_b,
                                          const float* W_ws, const float* b_ws, float* wsg){
  int b = blockIdx.x, tid = threadIdx.x;
  __shared__ float sh[512];
  __shared__ float hs[1024];
  const float* g = gates + (size_t)b*G4H;
  float hv[4]; float s = 0.0f, q = 0.0f;
  #pragma unroll
  for(int i=0;i<4;i++){
    int j = tid + i*256;
    float gi = sigm_(g[j]), gf = sigm_(g[1024+j]);
    float gg = tanhf(g[2048+j]), go = sigm_(g[3072+j]);
    float cn = gf * c[b*HH + j] + gi * gg;
    c[b*HH + j] = cn;
    float hhv = go * tanhf(cn);
    h[b*HH + j] = hhv;
    hv[i] = hhv; s += hhv; q += hhv*hhv;
  }
  block_red2(s, q, sh);
  float mu = s * (1.0f/1024.0f);
  float rs = rsqrtf(q * (1.0f/1024.0f) - mu*mu + 1e-5f);
  #pragma unroll
  for(int i=0;i<4;i++){
    int j = tid + i*256;
    float hnv = (hv[i]-mu)*rs*cn_g[j] + cn_b[j];
    hn[b*HH + j] = hnv; hs[j] = hnv;
  }
  __syncthreads();
  int head = tid >> 6, l = tid & 63;
  float p = 0.0f;
  for(int j = l; j < 1024; j += 64) p += hs[j] * W_ws[head*1024 + j];
  sh[tid] = p; __syncthreads();
  if(tid < 4){
    float acc = 0.0f;
    for(int i=0;i<64;i++) acc += sh[tid*64 + i];
    wsg[b*4 + tid] = sigm_(acc + b_ws[tid]);
  }
}

/* kD: heads GEMM (M=64, K=1024, N=768 = [rk|wk|ev]) + per-head LN / sigmoid */
__global__ __launch_bounds__(256) void kD(const float* hn, const __bf16* Wbh,
                                          const float* b_rk, const float* b_wk, const float* b_ev,
                                          const float* rkn_g, const float* rkn_b,
                                          const float* wkn_g, const float* wkn_b,
                                          float* rk_n, float* wk_n, float* ev){
  __shared__ __bf16 As[64*40];
  __shared__ float Ts[64*68];
  int tid = threadIdx.x, lane = tid & 31, w = tid >> 5;
  int m0 = (w & 3)*16, nh = w >> 2;
  int ngBase = blockIdx.x * 64;
  v8f ca = zero8(), cb = zero8();
  for(int k0 = 0; k0 < HH; k0 += 32){
    stage_a64x32(hn, HH, k0, As, tid);
    __syncthreads();
    v16bf a  = frag_a(As, 40, m0, 0, lane);
    v16bf b0 = frag_b(Wbh, HH, ngBase + nh*32,      k0, lane);
    v16bf b1 = frag_b(Wbh, HH, ngBase + nh*32 + 16, k0, lane);
    ca = wmma_bf16(a, b0, ca);
    cb = wmma_bf16(a, b1, cb);
    __syncthreads();
  }
  int nA = nh*32 + (lane & 15), nB = nA + 16;
  int mo = m0 + ((lane >> 4) << 3);
  #pragma unroll
  for(int r=0;r<8;r++){ Ts[(mo+r)*68 + nA] = ca[r]; Ts[(mo+r)*68 + nB] = cb[r]; }
  __syncthreads();
  if(tid < 64){
    int m = tid;
    int typ = ngBase >> 8, rel = ngBase & 255;
    if(typ == 2){
      for(int j=0;j<64;j++) ev[m*256 + rel + j] = sigm_(Ts[m*68 + j] + b_ev[rel + j]);
    } else {
      const float* bias = (typ == 0) ? b_rk : b_wk;
      const float* gg   = (typ == 0) ? rkn_g : wkn_g;
      const float* bb   = (typ == 0) ? rkn_b : wkn_b;
      float* dst        = (typ == 0) ? rk_n : wk_n;
      float s = 0.0f, q = 0.0f; float vrow[64];
      for(int j=0;j<64;j++){ float x = Ts[m*68+j] + bias[rel+j]; vrow[j] = x; s += x; q += x*x; }
      float mu = s*(1.0f/64.0f), rs = rsqrtf(q*(1.0f/64.0f) - mu*mu + 1e-5f);
      for(int j=0;j<64;j++) dst[m*256 + rel + j] = (vrow[j]-mu)*rs*gg[j] + bb[j];
    }
  }
}

/* kF: whole transformer block + NTM addressing, one workgroup per batch element.
   LDS arena (bytes): Mf 0 | Mb 32768 | Qb 51200 | Kb 67584 | Vt 83968 | Sc 100352
   | Pb 167936 | Os 202752 | Xf 237568 | Xb 270336 | Sa 288768 | Sw 290816 | end 292864 */
__global__ __launch_bounds__(256) void kF(float* mem, const float* rk_n, const float* wk_n,
    const float* ev, const float* wsg, float* rc,
    const __bf16* Wq, const __bf16* Wo, const __bf16* W1, const __bf16* W2,
    const float* attn_b, const float* attn_ob, const float* l1_b, const float* l2_b,
    const float* tn1_g, const float* tn1_b, const float* tn2_g, const float* tn2_b,
    const float* mn_g, const float* mn_b){
  extern __shared__ char smarena[];
  float*  Mf = (float*)(smarena);
  __bf16* Mb = (__bf16*)(smarena + 32768);
  __bf16* Qb = (__bf16*)(smarena + 51200);
  __bf16* Kb = (__bf16*)(smarena + 67584);
  __bf16* Vt = (__bf16*)(smarena + 83968);
  float*  Sc = (float*)(smarena + 100352);
  __bf16* Pb = (__bf16*)(smarena + 167936);
  float*  Os = (float*)(smarena + 202752);
  float*  Xf = (float*)(smarena + 237568);
  __bf16* Xb = (__bf16*)(smarena + 270336);
  float*  Sa = (float*)(smarena + 288768);
  float*  Sw = (float*)(smarena + 290816);

  int b = blockIdx.x, tid = threadIdx.x, lane = tid & 31, w = tid >> 5;
  float* gmem = mem + (size_t)b * NN * HS;

  /* Ph0: global mem tile -> LDS Mf via Tensor Data Mover. 2D descriptor:
     4B elements, tensor 64x128 (stride 64), tile 64x128, LDS address 0 (== Mf,
     no static LDS in this kernel). Groups 2/3 + extra group unused (zero). */
#if __has_builtin(__builtin_amdgcn_tensor_load_to_lds)
  if(w == 0){
    unsigned long long ga = (unsigned long long)(size_t)gmem;
    v4u g0; g0[0] = 1u;                       /* count=1, is_restore=0 */
    g0[1] = 0u;                               /* lds_addr = 0 (Mf)     */
    g0[2] = (unsigned)(ga & 0xffffffffu);     /* global_addr[31:0]     */
    g0[3] = (unsigned)((ga >> 32) & 0x01ffffffu) | (2u << 30); /* addr[56:32] | type=2 */
    v8i g1;
    g1[0] = (int)(2u << 16);                  /* data_size = 4B        */
    g1[1] = (int)(64u << 16);                 /* tensor_dim0 = 64      */
    g1[2] = (int)(128u << 16);                /* tensor_dim1 = 128     */
    g1[3] = (int)(64u << 16);                 /* tile_dim0 = 64        */
    g1[4] = 128;                              /* tile_dim1 = 128       */
    g1[5] = 64;                               /* tensor_dim0_stride    */
    g1[6] = 0; g1[7] = 0;
    v4i gz4; gz4[0]=0; gz4[1]=0; gz4[2]=0; gz4[3]=0;
    v8i gz8; for(int i=0;i<8;i++) gz8[i]=0;
    __builtin_amdgcn_tensor_load_to_lds(g0, g1, gz4, gz4, gz8, 0);
    __builtin_amdgcn_s_wait_tensorcnt(0);
  }
  __syncthreads();
  for(int i = tid*4; i < 128*64; i += 1024){
    v4f f = *(const v4f*)(Mf + i);
    int m = i >> 6, k = i & 63;
    v4bf o; o[0]=(__bf16)f.x; o[1]=(__bf16)f.y; o[2]=(__bf16)f.z; o[3]=(__bf16)f.w;
    *(v4bf*)(&Mb[m*72 + k]) = o;
  }
#else
  for(int i = tid*4; i < 128*64; i += 1024){
    v4f f = *(const v4f*)(gmem + i);
    *(v4f*)(Mf + i) = f;
    int m = i >> 6, k = i & 63;
    v4bf o; o[0]=(__bf16)f.x; o[1]=(__bf16)f.y; o[2]=(__bf16)f.z; o[3]=(__bf16)f.w;
    *(v4bf*)(&Mb[m*72 + k]) = o;
  }
#endif
  __syncthreads();

  /* Ph1: QKV GEMM (M=128, N=192, K=64); scatter q*0.25 / k / v^T */
  for(int tt = w; tt < 96; tt += 8){
    int m0 = (tt & 7)*16, n0 = (tt >> 3)*16;
    v8f cc = zero8();
    for(int k0 = 0; k0 < 64; k0 += 32){
      v16bf a  = frag_a(Mb, 72, m0, k0, lane);
      v16bf bb = frag_b(Wq, 64, n0, k0, lane);
      cc = wmma_bf16(a, bb, cc);
    }
    int n = n0 + (lane & 15), mo = m0 + ((lane >> 4) << 3);
    float bias = attn_b[n];
    #pragma unroll
    for(int r=0;r<8;r++){
      float vv = cc[r] + bias; int m = mo + r;
      if(n < 64)       Qb[(n >> 4)*2048 + m*16 + (n & 15)] = (__bf16)(vv * 0.25f);
      else if(n < 128){ int nn = n - 64;  Kb[(nn >> 4)*2048 + m*16 + (nn & 15)] = (__bf16)vv; }
      else            { int nn = n - 128; Vt[(nn >> 4)*2048 + (nn & 15)*128 + m] = (__bf16)vv; }
    }
  }
  __syncthreads();

  /* Ph2: per-head attention */
  for(int hh = 0; hh < 4; hh++){
    for(int tt = w; tt < 64; tt += 8){             /* scores: K=16 zero-padded to 32 */
      int m0 = (tt & 7)*16, n0 = (tt >> 3)*16;
      v16bf a, bb;
      {
        const __bf16* p = Qb + hh*2048 + (m0 + (lane & 15))*16 + ((lane >> 4)*8);
        #pragma unroll
        for(int i=0;i<8;i++){ a[i] = p[i]; a[i+8] = (__bf16)0.0f; }
      }
      {
        const __bf16* p = Kb + hh*2048 + (n0 + (lane & 15))*16 + ((lane >> 4)*8);
        #pragma unroll
        for(int i=0;i<8;i++){ bb[i] = p[i]; bb[i+8] = (__bf16)0.0f; }
      }
      v8f cc = zero8();
      cc = wmma_bf16(a, bb, cc);
      int n = n0 + (lane & 15), mo = m0 + ((lane >> 4) << 3);
      #pragma unroll
      for(int r=0;r<8;r++) Sc[(mo+r)*132 + n] = cc[r];
    }
    __syncthreads();
    for(int m = tid; m < 128; m += 256){           /* softmax row -> Pb (bf16) */
      float mx = -1e30f;
      for(int j=0;j<128;j++) mx = fmaxf(mx, Sc[m*132 + j]);
      float ss = 0.0f;
      for(int j=0;j<128;j++){ float e = __expf(Sc[m*132 + j] - mx); Sc[m*132 + j] = e; ss += e; }
      float inv = 1.0f / ss;
      for(int j=0;j<128;j++) Pb[m*136 + j] = (__bf16)(Sc[m*132 + j] * inv);
    }
    __syncthreads();
    {                                              /* P @ V_h: M=128, N=16, K=128 */
      int m0 = w * 16;
      v8f cc = zero8();
      for(int k0 = 0; k0 < 128; k0 += 32){
        v16bf a  = frag_a(Pb, 136, m0, k0, lane);
        v16bf bb = frag_b(Vt + hh*2048, 128, 0, k0, lane);
        cc = wmma_bf16(a, bb, cc);
      }
      int d = lane & 15, mo = m0 + ((lane >> 4) << 3);
      #pragma unroll
      for(int r=0;r<8;r++) Os[(mo+r)*68 + hh*16 + d] = cc[r];
    }
    __syncthreads();
  }

  /* Ph3: out-proj + residual + LN(tn1) */
  __bf16* Ob = Mb;                                  /* reuse */
  for(int i = tid*4; i < 128*64; i += 1024){
    int m = i >> 6, k = i & 63;
    v4f f = *(const v4f*)(Os + m*68 + k);
    v4bf o; o[0]=(__bf16)f.x; o[1]=(__bf16)f.y; o[2]=(__bf16)f.z; o[3]=(__bf16)f.w;
    *(v4bf*)(&Ob[m*72 + k]) = o;
  }
  __syncthreads();
  for(int tt = w; tt < 32; tt += 8){
    int m0 = (tt & 7)*16, n0 = (tt >> 3)*16;
    v8f cc = zero8();
    for(int k0 = 0; k0 < 64; k0 += 32){
      v16bf a  = frag_a(Ob, 72, m0, k0, lane);
      v16bf bb = frag_b(Wo, 64, n0, k0, lane);
      cc = wmma_bf16(a, bb, cc);
    }
    int n = n0 + (lane & 15), mo = m0 + ((lane >> 4) << 3);
    #pragma unroll
    for(int r=0;r<8;r++){ int m = mo + r; Xf[m*64 + n] = Mf[m*64 + n] + cc[r] + attn_ob[n]; }
  }
  __syncthreads();
  for(int m = tid; m < 128; m += 256){
    float s = 0.0f, q = 0.0f;
    for(int j=0;j<64;j++){ float x = Xf[m*64 + j]; s += x; q += x*x; }
    float mu = s*(1.0f/64.0f), rs = rsqrtf(q*(1.0f/64.0f) - mu*mu + 1e-5f);
    for(int j=0;j<64;j++){ float x = (Xf[m*64 + j]-mu)*rs*tn1_g[j] + tn1_b[j]; Xf[m*64 + j] = x; Xb[m*72 + j] = (__bf16)x; }
  }
  __syncthreads();

  /* Ph4: FF1 + gelu */
  __bf16* F1b = Pb;                                 /* reuse */
  for(int tt = w; tt < 64; tt += 8){
    int m0 = (tt & 7)*16, n0 = (tt >> 3)*16;
    v8f cc = zero8();
    for(int k0 = 0; k0 < 64; k0 += 32){
      v16bf a  = frag_a(Xb, 72, m0, k0, lane);
      v16bf bb = frag_b(W1, 64, n0, k0, lane);
      cc = wmma_bf16(a, bb, cc);
    }
    int n = n0 + (lane & 15), mo = m0 + ((lane >> 4) << 3);
    float bias = l1_b[n];
    #pragma unroll
    for(int r=0;r<8;r++) F1b[(mo+r)*136 + n] = (__bf16)gelu_(cc[r] + bias);
  }
  __syncthreads();

  /* Ph5: FF2 + residual + LN(tn2) -> Mf; mem_n -> Nn */
  for(int tt = w; tt < 32; tt += 8){
    int m0 = (tt & 7)*16, n0 = (tt >> 3)*16;
    v8f cc = zero8();
    for(int k0 = 0; k0 < 128; k0 += 32){
      v16bf a  = frag_a(F1b, 136, m0, k0, lane);
      v16bf bb = frag_b(W2, 128, n0, k0, lane);
      cc = wmma_bf16(a, bb, cc);
    }
    int n = n0 + (lane & 15), mo = m0 + ((lane >> 4) << 3);
    #pragma unroll
    for(int r=0;r<8;r++){ int m = mo + r; Mf[m*64 + n] = Xf[m*64 + n] + cc[r] + l2_b[n]; }
  }
  __syncthreads();
  float* Nn = Sc;                                   /* reuse as [128][64] */
  for(int m = tid; m < 128; m += 256){
    float s = 0.0f, q = 0.0f;
    for(int j=0;j<64;j++){ float x = Mf[m*64 + j]; s += x; q += x*x; }
    float mu = s*(1.0f/64.0f), rs = rsqrtf(q*(1.0f/64.0f) - mu*mu + 1e-5f);
    for(int j=0;j<64;j++){ float x = (Mf[m*64 + j]-mu)*rs*tn2_g[j] + tn2_b[j]; Mf[m*64 + j] = x; }
    s = 0.0f; q = 0.0f;
    for(int j=0;j<64;j++){ float x = Mf[m*64 + j]; s += x; q += x*x; }
    mu = s*(1.0f/64.0f); rs = rsqrtf(q*(1.0f/64.0f) - mu*mu + 1e-5f);
    for(int j=0;j<64;j++) Nn[m*64 + j] = (Mf[m*64 + j]-mu)*rs*mn_g[j] + mn_b[j];
  }
  __syncthreads();

  /* Ph6: NTM addressing */
  for(int idx = tid; idx < 512; idx += 256){
    int hh = idx >> 7, m = idx & 127;
    const float* rv = rk_n + b*256 + hh*64;
    const float* wv = wk_n + b*256 + hh*64;
    float s1 = 0.0f, s2 = 0.0f;
    for(int j=0;j<64;j++){ float nm = Nn[m*64 + j]; s1 += rv[j]*nm; s2 += wv[j]*nm; }
    Sa[hh*128 + m] = s1; Sw[hh*128 + m] = s2;
  }
  __syncthreads();
  if(tid < 8){
    float* arr = (tid < 4) ? (Sa + tid*128) : (Sw + (tid-4)*128);
    float mx = -1e30f; for(int j=0;j<128;j++) mx = fmaxf(mx, arr[j]);
    float ss = 0.0f;   for(int j=0;j<128;j++){ float e = __expf(arr[j]-mx); arr[j] = e; ss += e; }
    float inv = 1.0f/ss; for(int j=0;j<128;j++) arr[j] *= inv;
  }
  __syncthreads();
  {                                                 /* rc = rw @ mem */
    int hh = tid >> 6, j = tid & 63;
    float acc = 0.0f;
    for(int m=0;m<128;m++) acc += Sa[hh*128 + m] * Mf[m*64 + j];
    rc[b*256 + hh*64 + j] = acc;
  }
  for(int i = tid; i < 8192; i += 256){             /* erase + add -> global mem */
    int m = i >> 6, j = i & 63;
    float e = 0.0f, a = 0.0f;
    for(int n=0;n<4;n++){
      float wwv = Sw[n*128 + m];
      e += wwv * ev[b*256 + n*64 + j];
      a += wwv * wsg[b*4 + n] * wk_n[b*256 + n*64 + j];
    }
    gmem[i] = Mf[m*64 + j] * (1.0f - e) + a;
  }
}

/* kG0: po = LN(cat(hn, rc)) */
__global__ __launch_bounds__(256) void kG0(const float* hn, const float* rc,
                                           const float* po_g, const float* po_b, float* po){
  int b = blockIdx.x, tid = threadIdx.x;
  __shared__ float sh[512];
  float v[5]; float s = 0.0f, q = 0.0f;
  #pragma unroll
  for(int i=0;i<5;i++){
    int j = tid + i*256;
    float x = (j < 1024) ? hn[b*1024 + j] : rc[b*256 + (j - 1024)];
    v[i] = x; s += x; q += x*x;
  }
  block_red2(s, q, sh);
  float mu = s*(1.0f/1280.0f), rs = rsqrtf(q*(1.0f/1280.0f) - mu*mu + 1e-5f);
  #pragma unroll
  for(int i=0;i<5;i++){ int j = tid + i*256; po[b*POD + j] = (v[i]-mu)*rs*po_g[j] + po_b[j]; }
}

/* kG: logits = po @ W_proj^T + b -> d_out[:, t, :]  (M=64, N=1024, K=1280) */
__global__ __launch_bounds__(256) void kG(const float* po, const __bf16* Wbp,
                                          const float* b_proj, float* out, int t){
  __shared__ __bf16 As[64*40];
  int tid = threadIdx.x, lane = tid & 31, w = tid >> 5;
  int m0 = (w & 3)*16, nh = w >> 2;
  int ngBase = blockIdx.x * 64;
  v8f ca = zero8(), cb = zero8();
  for(int k0 = 0; k0 < POD; k0 += 32){
    stage_a64x32(po, POD, k0, As, tid);
    __syncthreads();
    __builtin_prefetch(Wbp + (size_t)(ngBase + nh*32)*POD + k0 + 128, 0, 3);
    v16bf a  = frag_a(As, 40, m0, 0, lane);
    v16bf b0 = frag_b(Wbp, POD, ngBase + nh*32,      k0, lane);
    v16bf b1 = frag_b(Wbp, POD, ngBase + nh*32 + 16, k0, lane);
    ca = wmma_bf16(a, b0, ca);
    cb = wmma_bf16(a, b1, cb);
    __syncthreads();
  }
  int nA = ngBase + nh*32 + (lane & 15), nB = nA + 16;
  int mo = m0 + ((lane >> 4) << 3);
  float bA = b_proj[nA], bBv = b_proj[nB];
  #pragma unroll
  for(int r=0;r<8;r++){
    int m = mo + r;
    out[(size_t)m*262144 + (size_t)t*1024 + nA] = ca[r] + bA;
    out[(size_t)m*262144 + (size_t)t*1024 + nB] = cb[r] + bBv;
  }
}

/* final copy of mem, h, c into d_out tail */
__global__ void kCopy(const float* mem, const float* h, const float* c, float* out){
  int i = blockIdx.x*256 + threadIdx.x;
  if(i < 524288)            out[16777216 + i] = mem[i];
  else if(i < 589824)       out[17301504 + (i - 524288)] = h[i - 524288];
  else if(i < 655360)       out[17367040 + (i - 589824)] = c[i - 589824];
}

extern "C" void kernel_launch(void* const* d_in, const int* in_sizes, int n_in,
                              void* d_out, int out_size, void* d_ws, size_t ws_size,
                              hipStream_t stream){
  const float* x_emb  = (const float*)d_in[0];
  const float* in_g   = (const float*)d_in[1];
  const float* in_b   = (const float*)d_in[2];
  const float* W_ih   = (const float*)d_in[3];
  const float* W_hh   = (const float*)d_in[4];
  const float* b_ih   = (const float*)d_in[5];
  const float* b_hh   = (const float*)d_in[6];
  const float* cn_g   = (const float*)d_in[7];
  const float* cn_b   = (const float*)d_in[8];
  const float* W_rk   = (const float*)d_in[9];
  const float* b_rk   = (const float*)d_in[10];
  const float* W_wk   = (const float*)d_in[11];
  const float* b_wk   = (const float*)d_in[12];
  const float* W_ws   = (const float*)d_in[13];
  const float* b_ws   = (const float*)d_in[14];
  const float* W_ev   = (const float*)d_in[15];
  const float* b_ev   = (const float*)d_in[16];
  const float* attn_w = (const float*)d_in[17];
  const float* attn_b = (const float*)d_in[18];
  const float* attn_ow= (const float*)d_in[19];
  const float* attn_ob= (const float*)d_in[20];
  const float* l1_w   = (const float*)d_in[21];
  const float* l1_b   = (const float*)d_in[22];
  const float* l2_w   = (const float*)d_in[23];
  const float* l2_b   = (const float*)d_in[24];
  const float* tn1_g  = (const float*)d_in[25];
  const float* tn1_b  = (const float*)d_in[26];
  const float* tn2_g  = (const float*)d_in[27];
  const float* tn2_b  = (const float*)d_in[28];
  const float* rkn_g  = (const float*)d_in[29];
  const float* rkn_b  = (const float*)d_in[30];
  const float* wkn_g  = (const float*)d_in[31];
  const float* wkn_b  = (const float*)d_in[32];
  const float* mn_g   = (const float*)d_in[33];
  const float* mn_b   = (const float*)d_in[34];
  const float* po_g   = (const float*)d_in[35];
  const float* po_b   = (const float*)d_in[36];
  const float* W_proj = (const float*)d_in[37];
  const float* b_proj = (const float*)d_in[38];

  char* ws = (char*)d_ws;
  float* h     = (float*)(ws + OFF_H);
  float* c     = (float*)(ws + OFF_C);
  float* mem   = (float*)(ws + OFF_MEM);
  float* rc    = (float*)(ws + OFF_RC);
  float* hn    = (float*)(ws + OFF_HN);
  float* inp   = (float*)(ws + OFF_INP);
  float* gates = (float*)(ws + OFF_GATES);
  float* rk_n  = (float*)(ws + OFF_RKN);
  float* wk_n  = (float*)(ws + OFF_WKN);
  float* ev    = (float*)(ws + OFF_EV);
  float* wsg   = (float*)(ws + OFF_WSG);
  float* po    = (float*)(ws + OFF_PO);
  __bf16* Wbg  = (__bf16*)(ws + OFF_WBG);
  __bf16* Wbh  = (__bf16*)(ws + OFF_WBH);
  __bf16* Wbp  = (__bf16*)(ws + OFF_WBP);
  __bf16* Wbq  = (__bf16*)(ws + OFF_WBQ);
  __bf16* Wbo  = (__bf16*)(ws + OFF_WBO);
  __bf16* Wb1  = (__bf16*)(ws + OFF_WB1);
  __bf16* Wb2  = (__bf16*)(ws + OFF_WB2);

  /* init state (h|c|mem|rc contiguous) and pre-convert weights to bf16 (L2-resident) */
  kZero<<<(671744 + 255)/256, 256, 0, stream>>>((float*)ws, 671744);
  kCvtGates<<<(G4H*GK/4 + 255)/256, 256, 0, stream>>>(W_ih, W_hh, Wbg);
  kCvtHeads<<<(768*HH/4 + 255)/256, 256, 0, stream>>>(W_rk, W_wk, W_ev, Wbh);
  kCvt<<<(1024*1280 + 255)/256, 256, 0, stream>>>(W_proj, Wbp, 1024*1280);
  kCvt<<<(192*64 + 255)/256, 256, 0, stream>>>(attn_w, Wbq, 192*64);
  kCvt<<<(64*64 + 255)/256, 256, 0, stream>>>(attn_ow, Wbo, 64*64);
  kCvt<<<(128*64 + 255)/256, 256, 0, stream>>>(l1_w, Wb1, 128*64);
  kCvt<<<(64*128 + 255)/256, 256, 0, stream>>>(l2_w, Wb2, 64*128);

  for(int t = 0; t < LL; t++){
    kA <<<64, 256, 0, stream>>>(x_emb, rc, in_g, in_b, inp, t);
    kB <<<64, 256, 0, stream>>>(inp, h, Wbg, b_ih, b_hh, gates);
    kC <<<64, 256, 0, stream>>>(gates, c, h, hn, cn_g, cn_b, W_ws, b_ws, wsg);
    kD <<<12, 256, 0, stream>>>(hn, Wbh, b_rk, b_wk, b_ev, rkn_g, rkn_b, wkn_g, wkn_b,
                                rk_n, wk_n, ev);
    kF <<<64, 256, 292864, stream>>>(mem, rk_n, wk_n, ev, wsg, rc,
                                     Wbq, Wbo, Wb1, Wb2,
                                     attn_b, attn_ob, l1_b, l2_b,
                                     tn1_g, tn1_b, tn2_g, tn2_b, mn_g, mn_b);
    kG0<<<64, 256, 0, stream>>>(hn, rc, po_g, po_b, po);
    kG <<<16, 256, 0, stream>>>(po, Wbp, b_proj, (float*)d_out, t);
  }
  kCopy<<<2560, 256, 0, stream>>>(mem, h, c, (float*)d_out);
}
